// StaticRNN_7653631721845
// MI455X (gfx1250) — matrix operand order repaired and checked by reading, not compiled, not run on hardware
//
#include <hip/hip_runtime.h>
#include <hip/hip_bf16.h>

// LSTM stack: B=64, T=256, D=H=1024, L=4.
// MI455X strategy: input projection hoisted into one big parallel WMMA GEMM
// (f16 in, f32 accumulate); sequential recurrence as a fused per-timestep
// kernel (h@U WMMA + gates + state update). f16 weights (64MB) stay resident
// in the 192MB L2. Both WMMA k-loops are software-pipelined (register double
// buffering) so the per-iteration loads overlap the previous WMMA burst
// instead of serializing behind s_wait_loadcnt 0.

typedef _Float16 v4h  __attribute__((ext_vector_type(4)));
typedef _Float16 v8h  __attribute__((ext_vector_type(8)));
typedef _Float16 v16h __attribute__((ext_vector_type(16)));
typedef float    v8f  __attribute__((ext_vector_type(8)));

#define B_   64
#define T_   256
#define D_   1024
#define H_   1024
#define L_   4
#define M_   (B_ * T_)   /* 16384 */
#define G4_  (4 * H_)    /* 4096  */

static __device__ __forceinline__ v16h cat8(v8h lo, v8h hi) {
  return __builtin_shufflevector(lo, hi, 0,1,2,3,4,5,6,7,8,9,10,11,12,13,14,15);
}

// A-tile 16x32 f16 (row-major source, row stride ld elements).
// ISA layout: lanes 0-15 -> row M=lane, K chunks {0..7, 16..23};
//             lanes 16-31 -> row M=lane-16, K chunks {8..15, 24..31}.
static __device__ __forceinline__ v16h load_a16x32(const _Float16* __restrict__ base,
                                                   int ld, int k0, int lane) {
  int row = lane & 15;
  const _Float16* p = base + (size_t)row * ld + k0 + ((lane >> 4) << 3);
  v8h lo = *(const v8h*)p;
  v8h hi = *(const v8h*)(p + 16);
  return cat8(lo, hi);
}

// B-tile 32x16 f16 from transposed weight (row-major [N][K], row stride ld).
// ISA layout: lanes 0-15 -> col N=lane, K chunks {0..7, 16..23}; lanes 16-31 -> col N=lane-16, K chunks {8..15, 24..31}.
static __device__ __forceinline__ v16h load_b32x16(const _Float16* __restrict__ base,
                                                   int ld, int k0, int lane) {
  int col = lane & 15;
  const _Float16* p = base + (size_t)col * ld + k0 + ((lane >> 4) << 3);
  v8h lo = *(const v8h*)p;
  v8h hi = *(const v8h*)(p + 16);
  return cat8(lo, hi);
}

static __device__ __forceinline__ float sigmoid_(float x) {
  return 1.0f / (1.0f + __expf(-x));
}
static __device__ __forceinline__ float tanh_(float x) {
  return 1.0f - 2.0f / (__expf(2.0f * x) + 1.0f);  // saturates correctly at +/-inf
}

// ---------------- f32 -> f16 convert (layer-0 input) ----------------
__global__ __launch_bounds__(256) void cvt_f32_f16_kernel(const float* __restrict__ in,
                                                          _Float16* __restrict__ out, int n4) {
  int i = blockIdx.x * blockDim.x + threadIdx.x;
  if (i >= n4) return;
  float4 v = ((const float4*)in)[i];
  v4h o = { (_Float16)v.x, (_Float16)v.y, (_Float16)v.z, (_Float16)v.w };
  ((v4h*)out)[i] = o;
}

// ------------- LDS-tiled transpose+convert: [z][R][C] f32 -> [z][C][R] f16 -------------
__global__ __launch_bounds__(256) void transpose_cvt_kernel(const float* __restrict__ in,
                                                            _Float16* __restrict__ out,
                                                            int R, int C) {
  __shared__ float tile[32][33];
  const float* inz  = in  + (size_t)blockIdx.z * R * C;
  _Float16*    outz = out + (size_t)blockIdx.z * R * C;
  int c0 = blockIdx.x << 5, r0 = blockIdx.y << 5;
  int tx = threadIdx.x, ty = threadIdx.y;
#pragma unroll
  for (int i = 0; i < 32; i += 8)
    tile[ty + i][tx] = inz[(size_t)(r0 + ty + i) * C + c0 + tx];
  __syncthreads();
#pragma unroll
  for (int i = 0; i < 32; i += 8)
    outz[(size_t)(c0 + ty + i) * R + r0 + tx] = (_Float16)tile[tx][ty + i];
}

__global__ __launch_bounds__(256) void zero_kernel(uint32_t* __restrict__ p, int n) {
  int i = blockIdx.x * blockDim.x + threadIdx.x;
  if (i < n) p[i] = 0u;
}

// ---------------- xz = x @ W  (parallel, full-sequence), f16 out ----------------
// Wave computes a 16x64 tile: 1 A-load feeds 4 WMMAs per k-step.
// k-loop is software-pipelined: next iteration's tiles are loaded before the
// current iteration's WMMAs so global latency overlaps the matrix ops.
__global__ __launch_bounds__(256) void wmma_xproj_kernel(const _Float16* __restrict__ X,   // [M_,D_]
                                                         const _Float16* __restrict__ Wt,  // [G4_,D_]
                                                         _Float16* __restrict__ XZ) {      // [M_,G4_]
  int lane = threadIdx.x;
  int m0 = (blockIdx.y * 8 + threadIdx.y) << 4;
  int n0 = blockIdx.x << 6;
  const _Float16* xrow = X + (size_t)m0 * D_;
  const _Float16* w0 = Wt + (size_t)(n0 +  0) * D_;
  const _Float16* w1 = Wt + (size_t)(n0 + 16) * D_;
  const _Float16* w2 = Wt + (size_t)(n0 + 32) * D_;
  const _Float16* w3 = Wt + (size_t)(n0 + 48) * D_;
  v8f a0 = {}, a1 = {}, a2 = {}, a3 = {};

  v16h a  = load_a16x32(xrow, D_, 0, lane);
  v16h b0 = load_b32x16(w0, D_, 0, lane);
  v16h b1 = load_b32x16(w1, D_, 0, lane);
  v16h b2 = load_b32x16(w2, D_, 0, lane);
  v16h b3 = load_b32x16(w3, D_, 0, lane);
  for (int k = 0; k < D_ - 32; k += 32) {
    int kn = k + 32;
    v16h an  = load_a16x32(xrow, D_, kn, lane);
    v16h bn0 = load_b32x16(w0, D_, kn, lane);
    v16h bn1 = load_b32x16(w1, D_, kn, lane);
    v16h bn2 = load_b32x16(w2, D_, kn, lane);
    v16h bn3 = load_b32x16(w3, D_, kn, lane);
    a0 = __builtin_amdgcn_wmma_f32_16x16x32_f16(false, a, false, b0, (short)0, a0, false, false);
    a1 = __builtin_amdgcn_wmma_f32_16x16x32_f16(false, a, false, b1, (short)0, a1, false, false);
    a2 = __builtin_amdgcn_wmma_f32_16x16x32_f16(false, a, false, b2, (short)0, a2, false, false);
    a3 = __builtin_amdgcn_wmma_f32_16x16x32_f16(false, a, false, b3, (short)0, a3, false, false);
    a = an; b0 = bn0; b1 = bn1; b2 = bn2; b3 = bn3;
  }
  a0 = __builtin_amdgcn_wmma_f32_16x16x32_f16(false, a, false, b0, (short)0, a0, false, false);
  a1 = __builtin_amdgcn_wmma_f32_16x16x32_f16(false, a, false, b1, (short)0, a1, false, false);
  a2 = __builtin_amdgcn_wmma_f32_16x16x32_f16(false, a, false, b2, (short)0, a2, false, false);
  a3 = __builtin_amdgcn_wmma_f32_16x16x32_f16(false, a, false, b3, (short)0, a3, false, false);

  int roff = (lane >> 4) << 3;
  int col  = lane & 15;
#pragma unroll
  for (int r = 0; r < 8; ++r) {
    size_t o = (size_t)(m0 + roff + r) * G4_ + n0 + col;
    XZ[o +  0] = (_Float16)a0[r];
    XZ[o + 16] = (_Float16)a1[r];
    XZ[o + 32] = (_Float16)a2[r];
    XZ[o + 48] = (_Float16)a3[r];
  }
}

// ---------------- fused recurrent step: z = xz + h@U + b; gates; c,h update ----------------
// 256 waves: wave = (batch tile 0..3, hidden tile 0..63); each wave does the
// 4 gate tiles (i,f,g,o) sharing one A-load of h per k-step. Software-
// pipelined k-loop (this is the sequential critical path: only ~2 waves/SIMD
// are resident, so latency must be hidden inside the wave).
__global__ __launch_bounds__(256) void lstm_step_kernel(const _Float16* __restrict__ XZ,     // [M_,G4_]
                                                        const _Float16* __restrict__ Ut,     // [G4_,H_]
                                                        const float* __restrict__ bias,      // [G4_]
                                                        const _Float16* __restrict__ h_prev, // [B_,H_]
                                                        _Float16* __restrict__ h_new,        // [B_,H_]
                                                        float* __restrict__ c,               // [B_,H_]
                                                        _Float16* __restrict__ seq16,        // [B_,T_,H_]
                                                        float* __restrict__ seq32,           // d_out
                                                        int t, int write32) {
  int lane = threadIdx.x;
  int wid  = blockIdx.x * 8 + threadIdx.y;  // 0..255
  int m0 = (wid >> 6) << 4;                 // batch rows
  int n0 = (wid & 63) << 4;                 // hidden cols
  int roff = (lane >> 4) << 3;
  int colL = lane & 15;
  int col  = n0 + colL;

  // warm caches for the epilogue's xz lines while the GEMM runs
  {
    size_t zr = ((size_t)(m0 + roff) * T_ + t) * (size_t)G4_;
    __builtin_prefetch(XZ + zr + 0 * H_ + col, 0, 0);
    __builtin_prefetch(XZ + zr + 1 * H_ + col, 0, 0);
    __builtin_prefetch(XZ + zr + 2 * H_ + col, 0, 0);
    __builtin_prefetch(XZ + zr + 3 * H_ + col, 0, 0);
  }

  const _Float16* hrow = h_prev + (size_t)m0 * H_;
  const _Float16* u_i = Ut + (size_t)(0 * H_ + n0) * H_;
  const _Float16* u_f = Ut + (size_t)(1 * H_ + n0) * H_;
  const _Float16* u_g = Ut + (size_t)(2 * H_ + n0) * H_;
  const _Float16* u_o = Ut + (size_t)(3 * H_ + n0) * H_;
  v8f zi = {}, zf = {}, zg = {}, zo = {};

  v16h a  = load_a16x32(hrow, H_, 0, lane);
  v16h bi = load_b32x16(u_i, H_, 0, lane);
  v16h bf = load_b32x16(u_f, H_, 0, lane);
  v16h bg = load_b32x16(u_g, H_, 0, lane);
  v16h bo = load_b32x16(u_o, H_, 0, lane);
  for (int k = 0; k < H_ - 32; k += 32) {
    int kn = k + 32;
    v16h an  = load_a16x32(hrow, H_, kn, lane);
    v16h bni = load_b32x16(u_i, H_, kn, lane);
    v16h bnf = load_b32x16(u_f, H_, kn, lane);
    v16h bng = load_b32x16(u_g, H_, kn, lane);
    v16h bno = load_b32x16(u_o, H_, kn, lane);
    zi = __builtin_amdgcn_wmma_f32_16x16x32_f16(false, a, false, bi, (short)0, zi, false, false);
    zf = __builtin_amdgcn_wmma_f32_16x16x32_f16(false, a, false, bf, (short)0, zf, false, false);
    zg = __builtin_amdgcn_wmma_f32_16x16x32_f16(false, a, false, bg, (short)0, zg, false, false);
    zo = __builtin_amdgcn_wmma_f32_16x16x32_f16(false, a, false, bo, (short)0, zo, false, false);
    a = an; bi = bni; bf = bnf; bg = bng; bo = bno;
  }
  zi = __builtin_amdgcn_wmma_f32_16x16x32_f16(false, a, false, bi, (short)0, zi, false, false);
  zf = __builtin_amdgcn_wmma_f32_16x16x32_f16(false, a, false, bf, (short)0, zf, false, false);
  zg = __builtin_amdgcn_wmma_f32_16x16x32_f16(false, a, false, bg, (short)0, zg, false, false);
  zo = __builtin_amdgcn_wmma_f32_16x16x32_f16(false, a, false, bo, (short)0, zo, false, false);

  float bi_ = bias[0 * H_ + col], bf_ = bias[1 * H_ + col];
  float bg_ = bias[2 * H_ + col], bo_ = bias[3 * H_ + col];
#pragma unroll
  for (int r = 0; r < 8; ++r) {
    int b = m0 + roff + r;
    size_t zrow = ((size_t)b * T_ + t) * (size_t)G4_;
    float vi = zi[r] + (float)XZ[zrow + 0 * H_ + col] + bi_;
    float vf = zf[r] + (float)XZ[zrow + 1 * H_ + col] + bf_;
    float vg = zg[r] + (float)XZ[zrow + 2 * H_ + col] + bg_;
    float vo = zo[r] + (float)XZ[zrow + 3 * H_ + col] + bo_;
    float iv = sigmoid_(vi), fv = sigmoid_(vf), gv = tanh_(vg), ov = sigmoid_(vo);
    size_t s = (size_t)b * H_ + col;
    float cn = fv * c[s] + iv * gv;
    c[s] = cn;
    float hv = ov * tanh_(cn);
    h_new[s] = (_Float16)hv;
    size_t q = ((size_t)b * T_ + t) * (size_t)H_ + col;
    seq16[q] = (_Float16)hv;
    if (write32) seq32[q] = hv;
  }
}

extern "C" void kernel_launch(void* const* d_in, const int* in_sizes, int n_in,
                              void* d_out, int out_size, void* d_ws, size_t ws_size,
                              hipStream_t stream) {
  (void)in_sizes; (void)n_in; (void)out_size; (void)ws_size;
  const float* x  = (const float*)d_in[0];   // [B,T,D]
  const float* Ws = (const float*)d_in[1];   // [L,D,4H]
  const float* Us = (const float*)d_in[2];   // [L,H,4H]
  const float* bs = (const float*)d_in[3];   // [L,4H]
  float* out = (float*)d_out;                // [B,T,H] f32

  // Scratch layout (~256.5 MB):
  char* ws = (char*)d_ws;
  const size_t SEQ16 = (size_t)M_ * H_ * sizeof(_Float16);       // 32 MB
  _Float16* bufA = (_Float16*)ws;                                // layer input/output ping
  _Float16* bufB = (_Float16*)(ws + SEQ16);                      // ...pong
  _Float16* XZ   = (_Float16*)(ws + 2 * SEQ16);                  // 128 MB
  _Float16* Wt   = (_Float16*)(ws + 2 * SEQ16 + (size_t)M_ * G4_ * sizeof(_Float16));
  _Float16* Ut   = Wt + (size_t)L_ * G4_ * D_;                   // 32 MB each
  _Float16* h0   = Ut + (size_t)L_ * G4_ * H_;
  _Float16* h1   = h0 + (size_t)B_ * H_;
  float*    cst  = (float*)(h1 + (size_t)B_ * H_);

  dim3 wgp(32, 8);

  // 1) convert layer-0 input to f16
  {
    int n4 = M_ * D_ / 4;
    cvt_f32_f16_kernel<<<(n4 + 255) / 256, 256, 0, stream>>>(x, bufA, n4);
  }
  // 2) transpose+convert all weights once: [K,4H] f32 -> [4H,K] f16
  transpose_cvt_kernel<<<dim3(G4_ / 32, D_ / 32, L_), wgp, 0, stream>>>(Ws, Wt, D_, G4_);
  transpose_cvt_kernel<<<dim3(G4_ / 32, H_ / 32, L_), wgp, 0, stream>>>(Us, Ut, H_, G4_);

  const int state_words = (int)((2 * (size_t)B_ * H_ * sizeof(_Float16) +
                                 (size_t)B_ * H_ * sizeof(float)) / 4);  // h0,h1,c

  for (int l = 0; l < L_; ++l) {
    _Float16* inbuf  = (l & 1) ? bufB : bufA;
    _Float16* outbuf = (l & 1) ? bufA : bufB;
    // 3) parallel input projection for the whole sequence
    wmma_xproj_kernel<<<dim3(G4_ / 64, M_ / 128), wgp, 0, stream>>>(
        inbuf, Wt + (size_t)l * G4_ * D_, XZ);
    // 4) reset h/c state
    zero_kernel<<<(state_words + 255) / 256, 256, 0, stream>>>((uint32_t*)h0, state_words);
    // 5) sequential recurrence (ping-pong h state to avoid intra-kernel races)
    for (int t = 0; t < T_; ++t) {
      const _Float16* hp = (t & 1) ? h1 : h0;
      _Float16*       hn = (t & 1) ? h0 : h1;
      lstm_step_kernel<<<32, wgp, 0, stream>>>(
          XZ, Ut + (size_t)l * G4_ * H_, bs + (size_t)l * G4_,
          hp, hn, cst, outbuf, out, t, (l == L_ - 1) ? 1 : 0);
    }
  }
}
